// BiMambaBlock_352187318777
// MI455X (gfx1250) — matrix operand order repaired and checked by reading, not compiled, not run on hardware
//
#include <hip/hip_runtime.h>
#include <hip/hip_bf16.h>

// ---------------- problem constants ----------------
#define DMODEL   1024
#define DSTATE   16
#define DCONVW   4
#define DINNER   2048
#define DTRANK   64
#define BATCH    2
#define SEQLEN   2048
#define MROWS    (BATCH * SEQLEN)      // 4096 tokens

typedef __hip_bfloat16 bf16_t;
typedef __attribute__((ext_vector_type(16))) __bf16 v16bf;
typedef __attribute__((ext_vector_type(8)))  float  v8f;

// ---------------- f32 -> bf16 convert ----------------
__global__ void k_f32_to_bf16(const float* __restrict__ in, bf16_t* __restrict__ out, int n) {
    int i = blockIdx.x * blockDim.x + threadIdx.x;
    if (i < n) out[i] = __float2bfloat16(in[i]);
}

// ---------------- WMMA fragment load helper ----------------
// Loads 32 bytes (16 bf16) as two 16B chunks; off2 = element offset of 2nd chunk.
__device__ __forceinline__ v16bf ldfrag(const bf16_t* p, int off2) {
    union { v16bf v; uint4 u[2]; } f;
    f.u[0] = *reinterpret_cast<const uint4*>(p);
    f.u[1] = *reinterpret_cast<const uint4*>(p + off2);
    return f.v;
}

// ---------------- WMMA GEMM: C[M,N] = A[M,K] * W[N,K]^T ----------------
// Block: 128 threads (4 waves stacked in M). Wave tile: 32(M) x 16*NT(N).
// Per K=32 step: 2 A-frag + NT B-frag loads feed 2*NT WMMAs.
// MODE 0: Cf=v   1: Cb=bf16(v)   2: Cf=softplus(v+bias[n])   3: Cf=v+bias[n]
template<int NT, int MODE>
__global__ __launch_bounds__(128)
void k_gemm_wmma(const bf16_t* __restrict__ A, const bf16_t* __restrict__ W,
                 float* __restrict__ Cf, bf16_t* __restrict__ Cb,
                 const float* __restrict__ bias, int K, int ldc) {
    const int wave = threadIdx.x >> 5;
    const int lane = threadIdx.x & 31;
    const int m0 = blockIdx.x * 128 + wave * 32;
    const int n0 = blockIdx.y * (16 * NT);

    // A-fragment (16x32 bf16): lane<16 -> row=lane, K {0..7}+{16..23}; lane>=16 -> K {8..15}+{24..31}
    const bf16_t* Ap0 = A + (size_t)(m0 + (lane & 15)) * K + ((lane >> 4) << 3);
    const bf16_t* Ap1 = Ap0 + (size_t)16 * K;
    // B-fragment (32x16): lane = column, lane<16 -> K {0..7}+{16..23}, lane>=16 -> K {8..15}+{24..31}
    const bf16_t* Bp[NT];
    #pragma unroll
    for (int j = 0; j < NT; ++j)
        Bp[j] = W + (size_t)(n0 + j * 16 + (lane & 15)) * K + ((lane >> 4) << 3);

    v8f acc0[NT], acc1[NT];
    #pragma unroll
    for (int j = 0; j < NT; ++j) { acc0[j] = {}; acc1[j] = {}; }

    for (int k = 0; k < K; k += 32) {
        v16bf a0 = ldfrag(Ap0 + k, 16);
        v16bf a1 = ldfrag(Ap1 + k, 16);
        #pragma unroll
        for (int j = 0; j < NT; ++j) {
            v16bf b = ldfrag(Bp[j] + k, 16);
            acc0[j] = __builtin_amdgcn_wmma_f32_16x16x32_bf16(false, a0, false, b, (short)0, acc0[j], false, false);
            acc1[j] = __builtin_amdgcn_wmma_f32_16x16x32_bf16(false, a1, false, b, (short)0, acc1[j], false, false);
        }
    }

    // C/D layout: element i of v8f -> M = i + 8*(lane>=16), N = lane&15
    const int col = lane & 15;
    const int mb  = (lane >> 4) << 3;
    #pragma unroll
    for (int j = 0; j < NT; ++j) {
        #pragma unroll
        for (int half = 0; half < 2; ++half) {
            const v8f& c = half ? acc1[j] : acc0[j];
            const int mm = m0 + half * 16 + mb;
            const int nn = n0 + j * 16 + col;
            #pragma unroll
            for (int i = 0; i < 8; ++i) {
                float v = c[i];
                size_t idx = (size_t)(mm + i) * ldc + nn;
                if (MODE == 0)       Cf[idx] = v;
                else if (MODE == 1)  Cb[idx] = __float2bfloat16(v);
                else if (MODE == 2) { float t = v + bias[nn];
                                      Cf[idx] = (t > 20.0f) ? t : log1pf(__expf(t)); }
                else                 Cf[idx] = v + bias[nn];
            }
        }
    }
}

// ---------------- depthwise causal conv + SiLU ----------------
// xz: [B*L, 2*DINNER] bf16 (xs in cols [0,DINNER)); u: [B*L, DINNER] bf16
// rev=0: u[t] = silu(b + sum_k w[k]*xs[t-3+k]); rev=1: reads xs[t+3-k]
__global__ void k_conv_silu(const bf16_t* __restrict__ xz, const float* __restrict__ w,
                            const float* __restrict__ bias, bf16_t* __restrict__ u, int rev) {
    int i = blockIdx.x * blockDim.x + threadIdx.x;        // (b*L+t)*DINNER + d
    if (i >= MROWS * DINNER) return;
    int d  = i & (DINNER - 1);
    int bt = i >> 11;
    int t  = bt % SEQLEN;
    int b  = bt / SEQLEN;
    float acc = bias[d];
    #pragma unroll
    for (int k = 0; k < DCONVW; ++k) {
        int tt = rev ? (t + (DCONVW - 1) - k) : (t - (DCONVW - 1) + k);
        if (tt >= 0 && tt < SEQLEN)
            acc += w[d * DCONVW + k] *
                   __bfloat162float(xz[((size_t)(b * SEQLEN + tt)) * (2 * DINNER) + d]);
    }
    float s = acc / (1.0f + __expf(-acc));
    u[i] = __float2bfloat16(s);
}

// ---------------- slice dt columns out of x_dbl, convert to bf16 ----------------
__global__ void k_split_dt(const float* __restrict__ x_dbl, bf16_t* __restrict__ dt, int total) {
    int i = blockIdx.x * blockDim.x + threadIdx.x;
    if (i >= total) return;
    int r = i >> 6, c = i & 63;
    dt[i] = __float2bfloat16(x_dbl[r * (DTRANK + 2 * DSTATE) + c]);
}

// ---------------- selective scan: one thread per (b, d) channel ----------------
__global__ void k_scan(const float* __restrict__ delta, const bf16_t* __restrict__ u,
                       const float* __restrict__ x_dbl, const float* __restrict__ A_log,
                       float* __restrict__ y, int rev) {
    int idx = blockIdx.x * blockDim.x + threadIdx.x;      // [0, BATCH*DINNER)
    if (idx >= BATCH * DINNER) return;
    int b = idx >> 11;
    int d = idx & (DINNER - 1);
    float A[DSTATE], h[DSTATE];
    #pragma unroll
    for (int s = 0; s < DSTATE; ++s) { A[s] = -__expf(A_log[d * DSTATE + s]); h[s] = 0.0f; }
    for (int step = 0; step < SEQLEN; ++step) {
        int t = rev ? (SEQLEN - 1 - step) : step;
        size_t base = (size_t)(b * SEQLEN + t);
        float dl = delta[base * DINNER + d];
        float uu = __bfloat162float(u[base * DINNER + d]);
        float du = dl * uu;
        const float* BC = x_dbl + base * (DTRANK + 2 * DSTATE);
        float yv = 0.0f;
        #pragma unroll
        for (int s = 0; s < DSTATE; ++s) {
            float dA = __expf(dl * A[s]);
            h[s] = h[s] * dA + du * BC[DTRANK + s];          // B_t
            yv  += h[s] * BC[DTRANK + DSTATE + s];           // C_t
        }
        y[base * DINNER + d] = yv;
    }
}

// ---------------- gate: (y + u*D) * silu(z) -> bf16 ----------------
__global__ void k_gate(const float* __restrict__ y, const bf16_t* __restrict__ u,
                       const float* __restrict__ Dv, const bf16_t* __restrict__ xz,
                       bf16_t* __restrict__ out) {
    int i = blockIdx.x * blockDim.x + threadIdx.x;
    if (i >= MROWS * DINNER) return;
    int d = i & (DINNER - 1);
    size_t row = (size_t)(i >> 11);
    float z = __bfloat162float(xz[row * (2 * DINNER) + DINNER + d]);
    float s = z / (1.0f + __expf(-z));
    float v = (y[i] + __bfloat162float(u[i]) * Dv[d]) * s;
    out[i] = __float2bfloat16(v);
}

// ---------------- host side ----------------
template<int NT>
static inline void gemm_nt(hipStream_t s, const bf16_t* A, const bf16_t* W,
                           float* Cf, bf16_t* Cb, const float* bias,
                           int M, int N, int K, int ldc, int mode) {
    dim3 g(M / 128, N / (16 * NT));
    switch (mode) {
    case 0: k_gemm_wmma<NT, 0><<<g, 128, 0, s>>>(A, W, Cf, Cb, bias, K, ldc); break;
    case 1: k_gemm_wmma<NT, 1><<<g, 128, 0, s>>>(A, W, Cf, Cb, bias, K, ldc); break;
    case 2: k_gemm_wmma<NT, 2><<<g, 128, 0, s>>>(A, W, Cf, Cb, bias, K, ldc); break;
    default:k_gemm_wmma<NT, 3><<<g, 128, 0, s>>>(A, W, Cf, Cb, bias, K, ldc); break;
    }
}

extern "C" void kernel_launch(void* const* d_in, const int* in_sizes, int n_in,
                              void* d_out, int out_size, void* d_ws, size_t ws_size,
                              hipStream_t stream) {
    const float* x        = (const float*)d_in[0];
    const float* fusion_w = (const float*)d_in[19];
    const float* fusion_b = (const float*)d_in[20];

    char*  ws  = (char*)d_ws;
    size_t off = 0;
    auto take = [&](size_t bytes) -> void* {
        void* p = ws + off; off = (off + bytes + 255) & ~(size_t)255; return p;
    };

    // bf16 copies of x and all weights
    bf16_t* x_bf = (bf16_t*)take((size_t)MROWS * DMODEL * 2);
    bf16_t *w_in[2], *w_xp[2], *w_dt[2], *w_op[2];
    for (int dir = 0; dir < 2; ++dir) {
        w_in[dir] = (bf16_t*)take((size_t)(2 * DINNER) * DMODEL * 2);
        w_xp[dir] = (bf16_t*)take((size_t)(DTRANK + 2 * DSTATE) * DINNER * 2);
        w_dt[dir] = (bf16_t*)take((size_t)DINNER * DTRANK * 2);
        w_op[dir] = (bf16_t*)take((size_t)DMODEL * DINNER * 2);
    }
    bf16_t* w_fu = (bf16_t*)take((size_t)DMODEL * (2 * DMODEL) * 2);   // [1024, 2048]

    // per-direction reusable activation buffers
    bf16_t* xz_bf = (bf16_t*)take((size_t)MROWS * 2 * DINNER * 2);
    bf16_t* u_bf  = (bf16_t*)take((size_t)MROWS * DINNER * 2);
    float*  x_dbl = (float*) take((size_t)MROWS * (DTRANK + 2 * DSTATE) * 4);
    bf16_t* dt_bf = (bf16_t*)take((size_t)MROWS * DTRANK * 2);
    float*  delta = (float*) take((size_t)MROWS * DINNER * 4);
    float*  y_buf = (float*) take((size_t)MROWS * DINNER * 4);
    bf16_t* yg_bf = (bf16_t*)take((size_t)MROWS * DINNER * 2);
    bf16_t* comb  = (bf16_t*)take((size_t)MROWS * 2 * DMODEL * 2);
    (void)ws_size; (void)in_sizes; (void)n_in; (void)out_size;

    auto cvt = [&](const float* src, bf16_t* dst, int n) {
        k_f32_to_bf16<<<(n + 255) / 256, 256, 0, stream>>>(src, dst, n);
    };

    cvt(x, x_bf, MROWS * DMODEL);
    for (int dir = 0; dir < 2; ++dir) {
        int base = 1 + dir * 9;
        cvt((const float*)d_in[base + 0], w_in[dir], 2 * DINNER * DMODEL);
        cvt((const float*)d_in[base + 3], w_xp[dir], (DTRANK + 2 * DSTATE) * DINNER);
        cvt((const float*)d_in[base + 4], w_dt[dir], DINNER * DTRANK);
        cvt((const float*)d_in[base + 8], w_op[dir], DMODEL * DINNER);
    }
    cvt(fusion_w, w_fu, DMODEL * 2 * DMODEL);

    const int nElem = MROWS * DINNER;             // 8.4M
    for (int dir = 0; dir < 2; ++dir) {
        int base = 1 + dir * 9;
        const float* conv_w    = (const float*)d_in[base + 1];
        const float* conv_b    = (const float*)d_in[base + 2];
        const float* dt_proj_b = (const float*)d_in[base + 5];
        const float* A_log     = (const float*)d_in[base + 6];
        const float* Dskip     = (const float*)d_in[base + 7];

        // xz = x @ in_proj^T         [4096, 4096] bf16
        gemm_nt<4>(stream, x_bf, w_in[dir], nullptr, xz_bf, nullptr,
                   MROWS, 2 * DINNER, DMODEL, 2 * DINNER, 1);
        // u = silu(causal depthwise conv(xs) + b)   (rev flips taps for bwd dir)
        k_conv_silu<<<(nElem + 255) / 256, 256, 0, stream>>>(xz_bf, conv_w, conv_b, u_bf, dir);
        // x_dbl = u @ x_proj^T       [4096, 96] f32  (N=96 -> 32-wide N tiles)
        gemm_nt<2>(stream, u_bf, w_xp[dir], x_dbl, nullptr, nullptr,
                   MROWS, DTRANK + 2 * DSTATE, DINNER, DTRANK + 2 * DSTATE, 0);
        // dt slice -> bf16
        k_split_dt<<<(MROWS * DTRANK + 255) / 256, 256, 0, stream>>>(x_dbl, dt_bf, MROWS * DTRANK);
        // delta = softplus(dt @ dt_proj^T + b)   [4096, 2048] f32
        gemm_nt<4>(stream, dt_bf, w_dt[dir], delta, nullptr, dt_proj_b,
                   MROWS, DINNER, DTRANK, DINNER, 2);
        // sequential selective scan (rev iterates time backwards for bwd dir)
        k_scan<<<(BATCH * DINNER + 255) / 256, 256, 0, stream>>>(delta, u_bf, x_dbl, A_log, y_buf, dir);
        // gate: (y + u*D) * silu(z)  -> bf16
        k_gate<<<(nElem + 255) / 256, 256, 0, stream>>>(y_buf, u_bf, Dskip, xz_bf, yg_bf);
        // out_proj written straight into combined[:, dir*1024 : dir*1024+1024]
        gemm_nt<4>(stream, yg_bf, w_op[dir], nullptr, comb + dir * DMODEL, nullptr,
                   MROWS, DMODEL, DINNER, 2 * DMODEL, 1);
    }

    // fused output: combined @ fusion_w^T + fusion_b -> d_out (f32)
    gemm_nt<4>(stream, comb, w_fu, (float*)d_out, nullptr, fusion_b,
               MROWS, DMODEL, 2 * DMODEL, DMODEL, 3);
}
